// STDF_simple_86165633893060
// MI455X (gfx1250) — compile-verified, matrix operand order checked by reading, not run on hardware
//
#include <hip/hip_runtime.h>

// ---------------------------------------------------------------------------
// STDF deformable-conv pipeline for MI455X (gfx1250, wave32, WMMA bf16).
//   B=2, C=64, H=W=128, nf=64, K9=9, om channels = 64*27 = 1728.
// Dominant cost: offset/mask conv (64 -> 1728 ch, ~65.5 GFLOP) -> implicit
// GEMM with v_wmma_f32_16x16x32_bf16. Activations kept NHWC bf16 so A
// fragments are contiguous 16B loads per half-lane (ISA 16-bit A layout).
// conv_om: one block per image row (128 px), 8 waves; each wave keeps one
// B-fragment in registers across 8 M-tile WMMAs (8x weight-traffic reuse);
// halo rows staged into LDS by the Tensor Data Mover (tensor_load_to_lds,
// s_wait_tensorcnt).
// Workspace layout (bytes, all 256-aligned; needs ~246 MB):
//   xb      bf16 NHWC        @ 0          (4 MB)
//   featb   bf16 NHWC        @ 4 MB       (4 MB)
//   featf   f32  NCHW        @ 8 MB       (8 MB)
//   off     f32  [B,1152,HW] @ 16 MB      (151 MB)
//   msk     f32  [B,576,HW]  @ 160 MB     (75.5 MB)
//   wpk_in  bf16 packed      @ 243,269,632  (73,728 B)
//   wpk_om  bf16 packed      @ 243,343,360  (1,990,656 B)
//   wpk_dc  bf16 packed      @ 245,334,016  (73,728 B)
// ---------------------------------------------------------------------------

typedef unsigned short u16;
typedef __attribute__((ext_vector_type(8)))  u16   u16x8;
typedef __attribute__((ext_vector_type(16))) u16   u16x16;
typedef __attribute__((ext_vector_type(16))) __bf16 bf16x16;
typedef __attribute__((ext_vector_type(8)))  float  f32x8;
typedef __attribute__((ext_vector_type(4)))  unsigned int u32x4;
typedef __attribute__((ext_vector_type(8)))  int    i32x8;
typedef __attribute__((ext_vector_type(4)))  int    i32x4;

#define HH 128
#define WW 128
#define HWPIX (HH * WW)
#define CIN 64
#define NF 64
#define OMCH 1728       // 64 * 27
#define OFFCH 1152      // 64 * 18
#define MSKCH 576       // 64 * 9
#define CK 576          // C * 9 (K-dim of final GEMM)

// ws offsets
#define OFF_XB     ((size_t)0)
#define OFF_FEATB  ((size_t)4194304)
#define OFF_FEATF  ((size_t)8388608)
#define OFF_OFF    ((size_t)16777216)
#define OFF_MSK    ((size_t)167772160)
#define OFF_WIN    ((size_t)243269632)
#define OFF_WOM    ((size_t)243343360)
#define OFF_WDC    ((size_t)245334016)

// fp32 -> bf16 (round-to-nearest-even)
__device__ __forceinline__ u16 f2bf(float f) {
    unsigned u = __builtin_bit_cast(unsigned, f);
    u = (u + 0x7FFFu + ((u >> 16) & 1u)) >> 16;
    return (u16)u;
}

__device__ __forceinline__ bf16x16 combine(u16x8 lo, u16x8 hi) {
    u16x16 t;
#pragma unroll
    for (int i = 0; i < 8; ++i) { t[i] = lo[i]; t[i + 8] = hi[i]; }
    return __builtin_bit_cast(bf16x16, t);
}

// ISA 16-bit A 16x32 layout: lane L holds row M = L%16; half = L/16;
// elements 0..7 -> K = 8*half+0..7 ; elements 8..15 -> K = 16+8*half+0..7.

// A-fragment from LDS halo tile smem[3][18][64] (conv_in staging).
__device__ __forceinline__ bf16x16 load_a_frag(const u16* smem, int t, int q, int lane) {
    int m = lane & 15, half = lane >> 4;
    int dy = t / 3, dx = t % 3;
    const u16* row = &smem[(dy * 18 + m + dx) * 64 + q * 32 + 8 * half];
    u16x8 a0 = *(const u16x8*)row;
    u16x8 a1 = *(const u16x8*)(row + 16);
    return combine(a0, a1);
}

// A-fragment from full-row LDS tile smem[3][128][64]; col may be -1..128,
// out-of-image columns produce zeros (predicated loads only; WMMA itself
// still executes with full EXEC).
__device__ __forceinline__ bf16x16 load_a_row(const u16* smem, int dy, int col,
                                              int q, int half) {
    u16x8 a0 = {}, a1 = {};
    if ((unsigned)col < (unsigned)WW) {
        const u16* rp = &smem[(dy * WW + col) * 64 + q * 32 + 8 * half];
        a0 = *(const u16x8*)rp;
        a1 = *(const u16x8*)(rp + 16);
    }
    return combine(a0, a1);
}

// Stage 3x18x64 halo tile (rows h-1..h+1, cols wbase-1..wbase+16, 64 ch).
__device__ __forceinline__ void stage_halo(u16* smem, const u16* in_nhwc,
                                           int b, int h, int wbase, int tid) {
    for (int idx = tid; idx < 432; idx += 128) {   // 432 x 16-byte chunks
        int c8 = (idx & 7) * 8;
        int cr = idx >> 3;               // 0..53
        int j = cr % 18, r = cr / 18;
        int y = h + r - 1, x = wbase + j - 1;
        u16x8 v = {};
        if ((unsigned)y < (unsigned)HH && (unsigned)x < (unsigned)WW)
            v = *(const u16x8*)&in_nhwc[(((size_t)b * HH + y) * WW + x) * 64 + c8];
        *(u16x8*)&smem[(r * 18 + j) * 64 + c8] = v;
    }
}

// ---------------- pack kernels ----------------------------------------------
// B-fragment layout (32x16 bf16, K x N): lane L holds column N = L%16; half =
// L/16; elements 0..7 -> K = 8*half+0..7 ; elements 8..15 -> K = 16+8*half+0..7.
__device__ __forceinline__ int kmap(int lane, int e) {
    int half = lane >> 4;
    return (e < 8) ? (8 * half + e) : (16 + 8 * half + (e - 8));
}

__global__ void stdf_pack_w3x3(const float* __restrict__ w, u16* __restrict__ pack, int total) {
    int idx = blockIdx.x * blockDim.x + threadIdx.x;
    if (idx >= total) return;
    int e = idx & 15;       int r = idx >> 4;
    int lane = r & 31;      r >>= 5;
    int q = r & 1;          r >>= 1;
    int t = r % 9;          int nt = r / 9;
    int c = q * 32 + kmap(lane, e);
    int n = nt * 16 + (lane & 15);
    int dy = t / 3, dx = t % 3;
    pack[idx] = f2bf(w[(((size_t)n * CIN + c) * 3 + dy) * 3 + dx]);
}

__global__ void stdf_pack_wdc(const float* __restrict__ w, u16* __restrict__ pack, int total) {
    int idx = blockIdx.x * blockDim.x + threadIdx.x;
    if (idx >= total) return;
    int e = idx & 15;       int r = idx >> 4;
    int lane = r & 31;      r >>= 5;
    int q = r % 18;         int nt = r / 18;
    int kg = q * 32 + kmap(lane, e);     // kg = c*9 + k9
    int c = kg / 9, k9 = kg % 9;
    int n = nt * 16 + (lane & 15);
    pack[idx] = f2bf(w[((size_t)n * CIN + c) * 9 + k9]);
}

__global__ void stdf_convert_x(const float* __restrict__ x, u16* __restrict__ xb) {
    int idx = blockIdx.x * blockDim.x + threadIdx.x;
    if (idx >= 2 * CIN * HWPIX) return;
    int b = idx / (CIN * HWPIX);
    int rem = idx % (CIN * HWPIX);
    int c = rem / HWPIX;
    int p = rem % HWPIX;
    int h = p / WW, w = p % WW;
    xb[(((size_t)b * HH + h) * WW + w) * 64 + c] = f2bf(x[idx]);
}

// ---------------- conv1: x -> feat (leaky relu 0.1) -------------------------
__global__ __launch_bounds__(128) void stdf_conv_in(
        const u16* __restrict__ xb, const u16* __restrict__ wpk,
        const float* __restrict__ bias,
        float* __restrict__ featf, u16* __restrict__ featb) {
    __shared__ u16 smem[3 * 18 * 64];
    int tid = threadIdx.x, lane = tid & 31, wave = tid >> 5;
    int tile = blockIdx.x;
    int wt = tile & 7;  int rest = tile >> 3;
    int h = rest % HH;  int b = rest / HH;
    int wbase = wt * 16;

    stage_halo(smem, xb, b, h, wbase, tid);
    __syncthreads();

    int nt = wave;                       // 4 waves x 16 out-ch = 64
    f32x8 acc = {};
#pragma unroll
    for (int t = 0; t < 9; ++t) {
#pragma unroll
        for (int q = 0; q < 2; ++q) {
            bf16x16 a = load_a_frag(smem, t, q, lane);
            const u16* wp = &wpk[(size_t)((((nt * 9 + t) * 2 + q) * 32 + lane)) * 16];
            bf16x16 bf = combine(*(const u16x8*)wp, *(const u16x8*)(wp + 8));
            acc = __builtin_amdgcn_wmma_f32_16x16x32_bf16(
                      false, a, false, bf, (short)0, acc, false, false);
        }
    }
    int nl = lane & 15, half = lane >> 4;
    int n = nt * 16 + nl;
    float bv = bias[n];
#pragma unroll
    for (int r = 0; r < 8; ++r) {
        int m = r + 8 * half;
        int w = wbase + m;
        float v = acc[r] + bv;
        v = v > 0.f ? v : 0.1f * v;                 // leaky_relu(0.1)
        featf[(((size_t)b * NF + n) * HH + h) * WW + w] = v;
        featb[(((size_t)b * HH + h) * WW + w) * 64 + n] = f2bf(v);
    }
}

// ---------------- conv2: feat -> offsets (f32) + sigmoid masks --------------
// One block per (b,h) image row: M = 128 pixels (8 M-tiles), N = 1728.
// Halo rows are DMA'd into LDS by the Tensor Data Mover: one 2-D tile,
// tile_dim0 = 8192 bf16 (one NHWC row = 16 KB contiguous), tile_dim1 = nrows,
// tensor_dim0_stride = 8192. Wave 0 issues it and waits on TENSORcnt.
__global__ __launch_bounds__(256) void stdf_conv_om(
        const u16* __restrict__ featb, const u16* __restrict__ wpk,
        const float* __restrict__ bias,
        float* __restrict__ off, float* __restrict__ msk) {
    __shared__ u16 smem[3 * WW * 64];    // 3 rows x 128 cols x 64 ch bf16 = 48 KB
    int tid = threadIdx.x, lane = tid & 31, wave = tid >> 5;
    int h = blockIdx.x % HH;
    int b = blockIdx.x / HH;

    // zero-fill out-of-image halo rows (only at h = 0 / 127)
    if (h == 0) {
        for (int i = tid; i < 1024; i += 256) *(u16x8*)&smem[i * 8] = (u16x8){};
    }
    if (h == HH - 1) {
        for (int i = tid; i < 1024; i += 256) *(u16x8*)&smem[2 * WW * 64 + i * 8] = (u16x8){};
    }

    // TDM: load valid rows [h-1+rstart .. h-1+rend] into LDS row slots.
    if (wave == 0) {
        int rstart = (h == 0) ? 1 : 0;
        int rend   = (h == HH - 1) ? 1 : 2;
        int nrows  = rend - rstart + 1;
        // generic LDS pointer: addr[31:0] is the LDS byte offset (flat aperture)
        unsigned lds0 = (unsigned)(size_t)&smem[rstart * WW * 64];
        unsigned long long ga =
            (unsigned long long)(size_t)(featb + (((size_t)b * HH + (h - 1 + rstart)) * WW) * 64);
        u32x4 g0;
        g0[0] = 1u;                                         // count=1 (valid), user mode
        g0[1] = lds0;                                       // lds_addr
        g0[2] = (unsigned)(ga & 0xFFFFFFFFu);               // global_addr[31:0]
        g0[3] = (unsigned)((ga >> 32) & 0x01FFFFFFu)        // global_addr[56:32]
              | (2u << 30);                                 // type = 2 (image)
        i32x8 g1;
        g1[0] = 1 << 16;                                    // data_size = 1 -> 2 bytes
        g1[1] = (int)(((unsigned)(WW * 64) & 0xFFFFu) << 16);   // tensor_dim0 lo16
        g1[2] = (int)(((unsigned)(WW * 64) >> 16)               // tensor_dim0 hi16
              | ((unsigned)nrows << 16));                       // tensor_dim1 lo16
        g1[3] = (int)(((unsigned)(WW * 64)) << 16);             // tile_dim0 = 8192
        g1[4] = nrows;                                          // tile_dim1 (tile_dim2=0)
        g1[5] = WW * 64;                                        // tensor_dim0_stride lo32
        g1[6] = 0;                                              // stride0 hi16 | stride1 lo16
        g1[7] = 0;                                              // stride1 hi32
        i32x4 gz4 = {0, 0, 0, 0};
        i32x8 gz8 = {0, 0, 0, 0, 0, 0, 0, 0};
        // 6-arg form (clang-23 / therock-10.0 headers): groups 0..3 + extra + cpol
        __builtin_amdgcn_tensor_load_to_lds(g0, g1, gz4, gz4, gz8, 0);
        __builtin_amdgcn_s_wait_tensorcnt(0);
    }
    __syncthreads();

    int nl = lane & 15, half = lane >> 4;
    for (int nt = wave; nt < OMCH / 16; nt += 8) {   // 13..14 n-tiles per wave
        f32x8 acc[8];
#pragma unroll
        for (int mt = 0; mt < 8; ++mt) acc[mt] = (f32x8){};
#pragma unroll
        for (int t = 0; t < 9; ++t) {
            int dy = t / 3, dx = t % 3;
#pragma unroll
            for (int q = 0; q < 2; ++q) {
                const u16* wp = &wpk[(size_t)((((nt * 9 + t) * 2 + q) * 32 + lane)) * 16];
                if (nt + 8 < OMCH / 16)
                    __builtin_prefetch(wp + 8 * 9 * 2 * 32 * 16, 0, 0);  // next weight tile
                bf16x16 bf = combine(*(const u16x8*)wp, *(const u16x8*)(wp + 8));
#pragma unroll
                for (int mt = 0; mt < 8; ++mt) {     // reuse B across 8 M-tiles
                    int col = mt * 16 + nl + dx - 1;
                    bf16x16 a = load_a_row(smem, dy, col, q, half);
                    acc[mt] = __builtin_amdgcn_wmma_f32_16x16x32_bf16(
                                  false, a, false, bf, (short)0, acc[mt], false, false);
                }
            }
        }
        int n = nt * 16 + nl;
        float bv = bias[n];
#pragma unroll
        for (int mt = 0; mt < 8; ++mt) {
#pragma unroll
            for (int r = 0; r < 8; ++r) {
                int w = mt * 16 + r + 8 * half;
                float v = acc[mt][r] + bv;
                int pix = h * WW + w;
                if (n < OFFCH) {
                    off[((size_t)b * OFFCH + n) * HWPIX + pix] = v;
                } else {
                    msk[((size_t)b * MSKCH + (n - OFFCH)) * HWPIX + pix] =
                        1.f / (1.f + __expf(-v));
                }
            }
        }
    }
}

// ---------------- kernel 3: deformable sampling + final GEMM + relu ---------
__global__ __launch_bounds__(128) void stdf_deform(
        const float* __restrict__ featf, const float* __restrict__ off,
        const float* __restrict__ msk, const u16* __restrict__ wdc_pk,
        const float* __restrict__ b_dc, float* __restrict__ out) {
    __shared__ u16 modl[16 * CK];        // 16 pixels x 576 modulated samples (bf16)
    int tid = threadIdx.x, lane = tid & 31, wave = tid >> 5;
    int tile = blockIdx.x;
    int wt = tile & 7;  int rest = tile >> 3;
    int h = rest % HH;  int b = rest / HH;
    int wbase = wt * 16;

    // phase 1: scattered bilinear sampling, modulated by mask, into LDS
    for (int i = tid; i < 16 * CK; i += 128) {       // 72 iters / thread
        int m = i & 15;
        int ck = i >> 4;                 // c*9 + k9
        int c = ck / 9, k9 = ck % 9;
        int wpix = wbase + m;
        int pix = h * WW + wpix;
        float oy = off[((size_t)b * OFFCH + (c * 18 + k9 * 2)) * HWPIX + pix];
        float ox = off[((size_t)b * OFFCH + (c * 18 + k9 * 2 + 1)) * HWPIX + pix];
        float mv = msk[((size_t)b * MSKCH + ck) * HWPIX + pix];
        float py = oy + (float)(k9 / 3 - 1) + (float)h;
        float px = ox + (float)(k9 % 3 - 1) + (float)wpix;
        float y0f = floorf(py), x0f = floorf(px);
        int y0 = (int)y0f, x0 = (int)x0f;
        float wy1 = py - y0f, wx1 = px - x0f;
        float wy0 = 1.f - wy1, wx0 = 1.f - wx1;
        const float* fp = featf + ((size_t)b * CIN + c) * HWPIX;
        auto g = [&](int yy, int xx) -> float {
            return ((unsigned)yy < (unsigned)HH && (unsigned)xx < (unsigned)WW)
                       ? fp[yy * WW + xx] : 0.f;
        };
        float s = wy0 * wx0 * g(y0, x0)     + wy0 * wx1 * g(y0, x0 + 1)
                + wy1 * wx0 * g(y0 + 1, x0) + wy1 * wx1 * g(y0 + 1, x0 + 1);
        modl[m * CK + ck] = f2bf(s * mv);
    }
    __syncthreads();

    // phase 2: [16 x 576] @ [576 x 64] GEMM, one 16-ch n-tile per wave
    int nt = wave;
    int m0 = lane & 15, half = lane >> 4;
    f32x8 acc = {};
#pragma unroll
    for (int q = 0; q < 18; ++q) {
        const u16* row = &modl[m0 * CK + q * 32 + 8 * half];
        bf16x16 a = combine(*(const u16x8*)row, *(const u16x8*)(row + 16));
        const u16* wp = &wdc_pk[(size_t)((nt * 18 + q) * 32 + lane) * 16];
        bf16x16 bf = combine(*(const u16x8*)wp, *(const u16x8*)(wp + 8));
        acc = __builtin_amdgcn_wmma_f32_16x16x32_bf16(
                  false, a, false, bf, (short)0, acc, false, false);
    }
    int n = nt * 16 + m0;
    float bv = b_dc[n];
#pragma unroll
    for (int r = 0; r < 8; ++r) {
        int m = r + 8 * half;
        float v = acc[r] + bv;
        out[(((size_t)b * NF + n) * HH + h) * WW + wbase + m] = v > 0.f ? v : 0.f;
    }
}

// ---------------------------------------------------------------------------
extern "C" void kernel_launch(void* const* d_in, const int* in_sizes, int n_in,
                              void* d_out, int out_size, void* d_ws, size_t ws_size,
                              hipStream_t stream) {
    const float* x    = (const float*)d_in[0];
    const float* w_in = (const float*)d_in[1];
    const float* b_in = (const float*)d_in[2];
    const float* w_om = (const float*)d_in[3];
    const float* b_om = (const float*)d_in[4];
    const float* w_dc = (const float*)d_in[5];
    const float* b_dc = (const float*)d_in[6];
    float* out = (float*)d_out;

    char* ws = (char*)d_ws;
    u16*   xb    = (u16*)  (ws + OFF_XB);
    u16*   featb = (u16*)  (ws + OFF_FEATB);
    float* featf = (float*)(ws + OFF_FEATF);
    float* off   = (float*)(ws + OFF_OFF);
    float* msk   = (float*)(ws + OFF_MSK);
    u16*   wpk_in = (u16*) (ws + OFF_WIN);
    u16*   wpk_om = (u16*) (ws + OFF_WOM);
    u16*   wpk_dc = (u16*) (ws + OFF_WDC);

    // weight packing + input conversion
    {
        int tot_in = (NF / 16) * 9 * 2 * 32 * 16;        // 36,864
        stdf_pack_w3x3<<<(tot_in + 255) / 256, 256, 0, stream>>>(w_in, wpk_in, tot_in);
        int tot_om = (OMCH / 16) * 9 * 2 * 32 * 16;      // 995,328
        stdf_pack_w3x3<<<(tot_om + 255) / 256, 256, 0, stream>>>(w_om, wpk_om, tot_om);
        int tot_dc = (NF / 16) * 18 * 32 * 16;           // 36,864
        stdf_pack_wdc<<<(tot_dc + 255) / 256, 256, 0, stream>>>(w_dc, wpk_dc, tot_dc);
        int tot_x = 2 * CIN * HWPIX;                     // 2,097,152
        stdf_convert_x<<<(tot_x + 255) / 256, 256, 0, stream>>>(x, xb);
    }

    int tiles = 2 * HH * (WW / 16);                      // 2048 pixel tiles
    stdf_conv_in<<<tiles, 128, 0, stream>>>(xb, wpk_in, b_in, featf, featb);
    stdf_conv_om<<<2 * HH, 256, 0, stream>>>(featb, wpk_om, b_om, off, msk);
    stdf_deform <<<tiles, 128, 0, stream>>>(featf, off, msk, wpk_dc, b_dc, out);
}
